// SelfAttention_70523363000423
// MI455X (gfx1250) — compile-verified, matrix operand order checked by reading, not run on hardware
//
#include <hip/hip_runtime.h>
#include <hip/hip_bf16.h>

// Problem constants (from reference)
constexpr int Bn = 4, Cn = 512, Hn = 64, Wn = 64;
constexpr int Ln = Hn * Wn;   // 4096
constexpr int CK = Cn / 8;    // 64
constexpr int CG = Cn / 2;    // 256

typedef _Float16 h16;
typedef __attribute__((ext_vector_type(8)))  _Float16 v8h;
typedef __attribute__((ext_vector_type(16))) _Float16 v16h;
typedef __attribute__((ext_vector_type(8)))  float    v8f;

__device__ __forceinline__ v8f vzero8() {
  v8f z;
#pragma unroll
  for (int i = 0; i < 8; ++i) z[i] = 0.0f;
  return z;
}

// Load a 16x32 (MxK) f16 A-style fragment from a row-major matrix
// (row stride = stride_h halfs). Per ISA 7.12.2: lane holds row lane&15,
// K-halves [kh*8, kh*8+7] and [16+kh*8, 16+kh*8+7] with kh = lane>>4.
// Two 16B loads -> b128. Also used for B fragments by loading B^T rows
// (B column-major mirrors A row-major).
__device__ __forceinline__ v16h frag_ld(const h16* base, int stride_h, int lane) {
  const int row = lane & 15;
  const h16* p = base + (size_t)row * stride_h + ((lane >> 4) << 3);
  v8h lo = *reinterpret_cast<const v8h*>(p);
  v8h hi = *reinterpret_cast<const v8h*>(p + 16);
  return __builtin_shufflevector(lo, hi, 0,1,2,3,4,5,6,7,8,9,10,11,12,13,14,15);
}

__device__ __forceinline__ v8f wmma16(v16h a, v16h b, v8f c) {
  // D = A(16x32 f16) * B(32x16 f16) + C(16x16 f32)
  return __builtin_amdgcn_wmma_f32_16x16x32_f16(false, a, false, b, (short)0, c,
                                                false, false);
}

// -------------------------------------------------------------------------
// K1: fused 1x1-conv projections. out(64 o x 64 l) tile per block, classic
// LDS-tiled f32 GEMM. grid.y selects {theta, phi, g[0..3]} channel group.
// theta/phi stored l-major (L x CK) f16; g stored TRANSPOSED c-major
// (CG x L) f16 so later WMMA B-fragments are contiguous along m.
// -------------------------------------------------------------------------
__global__ void __launch_bounds__(256) k_proj(
    const float* __restrict__ x,
    const float* __restrict__ Wt, const float* __restrict__ bt,
    const float* __restrict__ Wp, const float* __restrict__ bp,
    const float* __restrict__ Wg, const float* __restrict__ bg,
    h16* __restrict__ theta, h16* __restrict__ phi, h16* __restrict__ gT) {
  __shared__ float xs[32][65];
  __shared__ float wsm[64][33];
  const int b   = blockIdx.x >> 6;          // Ln/64 = 64 l-tiles per batch
  const int l0  = (blockIdx.x & 63) << 6;
  const int grp = blockIdx.y;
  const float* W; const float* bias; int o0 = 0;
  if (grp == 0)      { W = Wt; bias = bt; }
  else if (grp == 1) { W = Wp; bias = bp; }
  else               { W = Wg; bias = bg; o0 = (grp - 2) << 6; }
  const int tid = threadIdx.x;
  const int tl = (tid & 15) << 2;
  const int to = (tid >> 4) << 2;
  float acc[4][4] = {};
  for (int cc = 0; cc < Cn; cc += 32) {
#pragma unroll
    for (int i = 0; i < 8; ++i) {
      int e = tid + i * 256;
      xs[e >> 6][e & 63]  = x[((size_t)b * Cn + cc + (e >> 6)) * Ln + l0 + (e & 63)];
      wsm[e >> 5][e & 31] = W[(size_t)(o0 + (e >> 5)) * Cn + cc + (e & 31)];
    }
    __syncthreads();
#pragma unroll 8
    for (int k = 0; k < 32; ++k) {
      float av[4], xv[4];
#pragma unroll
      for (int i = 0; i < 4; ++i) av[i] = wsm[to + i][k];
#pragma unroll
      for (int j = 0; j < 4; ++j) xv[j] = xs[k][tl + j];
#pragma unroll
      for (int i = 0; i < 4; ++i)
#pragma unroll
        for (int j = 0; j < 4; ++j) acc[i][j] += av[i] * xv[j];
    }
    __syncthreads();
  }
#pragma unroll
  for (int i = 0; i < 4; ++i) {
#pragma unroll
    for (int j = 0; j < 4; ++j) {
      float v = acc[i][j] + bias[o0 + to + i];
      if (grp < 2) v = tanhf(v);
      if (grp == 0)
        theta[((size_t)b * Ln + l0 + tl + j) * CK + to + i] = (h16)v;
      else if (grp == 1)
        phi[((size_t)b * Ln + l0 + tl + j) * CK + to + i] = (h16)v;
      else
        gT[(size_t)b * CG * Ln + (size_t)(o0 + to + i) * Ln + l0 + tl + j] = (h16)v;
    }
  }
}

// K2: convert W_o (C x CG, row-major f32) to f16 once per call.
__global__ void __launch_bounds__(256) k_cvt_wo(const float* __restrict__ Wo,
                                                h16* __restrict__ WoH) {
  int i = blockIdx.x * 256 + threadIdx.x;
  if (i < Cn * CG) WoH[i] = (h16)Wo[i];
}

// -------------------------------------------------------------------------
// K3: column softmax statistics. softmax is over axis=1 (queries l), i.e.
// per KEY column m: colmax[m] = max_l S[l,m], colsum[m] = sum_l e^(S-max).
// One block per (b, 16-m tile); 4 waves stride over the 256 l-tiles with
// an online (max,sum) update; combine lane-halves via shfl_xor(16), then
// waves via LDS. Score tile = 2 x v_wmma_f32_16x16x32_f16 over K=CK=64.
// -------------------------------------------------------------------------
__global__ void __launch_bounds__(128) k_colstats(
    const h16* __restrict__ theta, const h16* __restrict__ phi,
    float* __restrict__ colmax, float* __restrict__ colsum) {
  const int b  = blockIdx.x >> 8;           // 256 m-tiles per batch
  const int m0 = (blockIdx.x & 255) << 4;
  const int wave = threadIdx.x >> 5, lane = threadIdx.x & 31;
  const h16* thb = theta + (size_t)b * Ln * CK;
  const h16* phb = phi   + (size_t)b * Ln * CK;
  const v16h pb0 = frag_ld(phb + (size_t)m0 * CK,      CK, lane);
  const v16h pb1 = frag_ld(phb + (size_t)m0 * CK + 32, CK, lane);
  float mx = -3.0e38f, sm = 0.0f;
  for (int lt = wave; lt < Ln / 16; lt += 4) {
    // branchless prefetch of next l-tile (wrap keeps the address valid;
    // prefetch is speculative + side-effect free, so no guard -> no exec
    // manipulation in the hot loop)
    __builtin_prefetch(
        thb + (size_t)((((lt + 4) & 255) << 4) + (lane & 15)) * CK, 0, 0);
    const h16* ab = thb + (size_t)(lt << 4) * CK;
    v16h a0 = frag_ld(ab,      CK, lane);
    v16h a1 = frag_ld(ab + 32, CK, lane);
    v8f s = vzero8();
    s = wmma16(a0, pb0, s);
    s = wmma16(a1, pb1, s);
    float tm = s[0];
#pragma unroll
    for (int r = 1; r < 8; ++r) tm = fmaxf(tm, s[r]);
    float nm = fmaxf(mx, tm);
    sm *= __expf(mx - nm);
#pragma unroll
    for (int r = 0; r < 8; ++r) sm += __expf(s[r] - nm);
    mx = nm;
  }
  // combine row-halves: column n lives in lanes n and n+16
  float mx2 = __shfl_xor(mx, 16, 32);
  float sm2 = __shfl_xor(sm, 16, 32);
  float nm = fmaxf(mx, mx2);
  sm = sm * __expf(mx - nm) + sm2 * __expf(mx2 - nm);
  mx = nm;
  __shared__ float smx[4][16], ssm[4][16];
  if (lane < 16) { smx[wave][lane] = mx; ssm[wave][lane] = sm; }
  __syncthreads();
  if (threadIdx.x < 16) {
    float M = smx[0][threadIdx.x], S = ssm[0][threadIdx.x];
#pragma unroll
    for (int w = 1; w < 4; ++w) {
      float m2 = smx[w][threadIdx.x], s2 = ssm[w][threadIdx.x];
      float n2 = fmaxf(M, m2);
      S = S * __expf(M - n2) + s2 * __expf(m2 - n2);
      M = n2;
    }
    colmax[(size_t)b * Ln + m0 + threadIdx.x] = M;
    colsum[(size_t)b * Ln + m0 + threadIdx.x] = S;
  }
}

// K4: fold 1/colsum into V (in place on gT; gT is regenerated each call).
__global__ void __launch_bounds__(256) k_norm_g(h16* __restrict__ gT,
                                                const float* __restrict__ colsum) {
  int i = blockIdx.x * 256 + threadIdx.x;   // B*CG*Ln = 4 * 2^20 elements
  int m = i & (Ln - 1);
  int b = i >> 20;                           // CG*Ln = 2^20
  gT[i] = (h16)((float)gT[i] / colsum[(size_t)b * Ln + m]);
}

// -------------------------------------------------------------------------
// K5: flash-style P*V. Each WAVE owns one 16-l query tile (so scores are
// computed exactly once per wave -- no intra-block redundancy); each wave
// accumulates 8 c-tiles (128 of the 256 channels, selected by blockIdx.y).
// Per 32-m chunk: 4 score WMMAs, exp(S - colmax) -> f16 via a per-wave
// 16x32 LDS tile (D-layout -> A-layout transpose; same-wave DS ops are
// in-order), then 8 P*V WMMAs against gn^T B-fragments.
// -------------------------------------------------------------------------
__global__ void __launch_bounds__(128) k_pv(
    const h16* __restrict__ theta, const h16* __restrict__ phi,
    const h16* __restrict__ gT, const float* __restrict__ colmax,
    h16* __restrict__ attnpre) {
  __shared__ alignas(16) h16 elds[4][16][32];
  const int b   = blockIdx.x >> 6;           // Ln/64 = 64 blocks per batch
  const int wave = threadIdx.x >> 5, lane = threadIdx.x & 31;
  const int l0  = ((blockIdx.x & 63) << 6) + (wave << 4);  // wave-private l-tile
  const int cbase = blockIdx.y << 7;         // c-half: 0 or 128
  const int n = lane & 15, hh = lane >> 4;
  const h16* thb = theta + (size_t)b * Ln * CK;
  const h16* phb = phi   + (size_t)b * Ln * CK;
  const h16* gb  = gT    + (size_t)b * CG * Ln;
  const float* cmb = colmax + (size_t)b * Ln;
  const v16h a0 = frag_ld(thb + (size_t)l0 * CK,      CK, lane);
  const v16h a1 = frag_ld(thb + (size_t)l0 * CK + 32, CK, lane);
  v8f acc[8];
#pragma unroll
  for (int ct = 0; ct < 8; ++ct) acc[ct] = vzero8();
  for (int m0 = 0; m0 < Ln; m0 += 32) {
    // branchless prefetch of one 128B phi row of the next m-chunk (wrapped)
    __builtin_prefetch(phb + (size_t)(((m0 + 32) & (Ln - 1)) + lane) * CK, 0, 0);
    v16h p00 = frag_ld(phb + (size_t)m0 * CK,             CK, lane);
    v16h p01 = frag_ld(phb + (size_t)m0 * CK + 32,        CK, lane);
    v16h p10 = frag_ld(phb + (size_t)(m0 + 16) * CK,      CK, lane);
    v16h p11 = frag_ld(phb + (size_t)(m0 + 16) * CK + 32, CK, lane);
    v8f s0 = vzero8(), s1 = vzero8();
    s0 = wmma16(a0, p00, s0); s0 = wmma16(a1, p01, s0);
    s1 = wmma16(a0, p10, s1); s1 = wmma16(a1, p11, s1);
    const float cm0 = cmb[m0 + n];
    const float cm1 = cmb[m0 + 16 + n];
#pragma unroll
    for (int r = 0; r < 8; ++r) {
      elds[wave][r + (hh << 3)][n]      = (h16)__expf(s0[r] - cm0);
      elds[wave][r + (hh << 3)][n + 16] = (h16)__expf(s1[r] - cm1);
    }
    v16h ea = frag_ld(&elds[wave][0][0], 32, lane);
#pragma unroll
    for (int ct = 0; ct < 8; ++ct) {
      const h16* gbase = gb + (size_t)(cbase + (ct << 4)) * Ln + m0;
      v16h gfrag = frag_ld(gbase, Ln, lane);
      acc[ct] = wmma16(ea, gfrag, acc[ct]);
    }
  }
  // store attn_pre (L x CG, l-major) as f16 for the Wo GEMM
#pragma unroll
  for (int ct = 0; ct < 8; ++ct)
#pragma unroll
    for (int r = 0; r < 8; ++r)
      attnpre[(size_t)b * Ln * CG + (size_t)(l0 + r + (hh << 3)) * CG +
              cbase + (ct << 4) + n] = (h16)acc[ct][r];
}

// -------------------------------------------------------------------------
// K6: Wo projection + alpha-blend. out[co,l] tile: A = Wo rows (co x c),
// B^T = attnpre rows (l x c). D-fragment lanes map to contiguous l ->
// coalesced f32 stores of both output halves.
// -------------------------------------------------------------------------
__global__ void __launch_bounds__(128) k_oproj(
    const h16* __restrict__ WoH, const h16* __restrict__ attnpre,
    const float* __restrict__ b_o, const float* __restrict__ gamma,
    const float* __restrict__ x, float* __restrict__ out1,
    float* __restrict__ out2) {
  const int b  = blockIdx.x >> 8;
  const int l0 = (blockIdx.x & 255) << 4;
  const int wave = threadIdx.x >> 5, lane = threadIdx.x & 31;
  const int n = lane & 15, hh = lane >> 4;
  const int co0 = (blockIdx.y << 6) + (wave << 4);
  const h16* apb = attnpre + (size_t)b * Ln * CG + (size_t)l0 * CG;
  v8f acc = vzero8();
#pragma unroll
  for (int k0 = 0; k0 < CG; k0 += 32) {
    v16h wa = frag_ld(WoH + (size_t)co0 * CG + k0, CG, lane);
    v16h ba = frag_ld(apb + k0, CG, lane);
    acc = wmma16(wa, ba, acc);
  }
  const float alpha = 1.0f / (1.0f + __expf(-gamma[0]));
  const int l = l0 + n;
#pragma unroll
  for (int r = 0; r < 8; ++r) {
    const int co = co0 + r + (hh << 3);
    const size_t idx = ((size_t)b * Cn + co) * Ln + l;
    const float y = acc[r] + b_o[co];
    out2[idx] = y;                                    // attn_g output
    out1[idx] = (1.0f - alpha) * x[idx] + alpha * y;  // blended output
  }
}

extern "C" void kernel_launch(void* const* d_in, const int* in_sizes, int n_in,
                              void* d_out, int out_size, void* d_ws, size_t ws_size,
                              hipStream_t stream) {
  (void)in_sizes; (void)n_in; (void)out_size; (void)ws_size;
  const float* x     = (const float*)d_in[0];
  const float* Wt    = (const float*)d_in[1];
  const float* bt    = (const float*)d_in[2];
  const float* Wp    = (const float*)d_in[3];
  const float* bp    = (const float*)d_in[4];
  const float* Wg    = (const float*)d_in[5];
  const float* bg    = (const float*)d_in[6];
  const float* Wo    = (const float*)d_in[7];
  const float* bo    = (const float*)d_in[8];
  const float* gamma = (const float*)d_in[9];

  float* out1 = (float*)d_out;
  float* out2 = out1 + (size_t)Bn * Cn * Ln;

  // Workspace layout (~20.4 MB, L2-resident)
  char* ws = (char*)d_ws;
  size_t off = 0;
  h16* theta   = (h16*)(ws + off); off += (size_t)Bn * Ln * CK * 2;  // 2 MB
  h16* phi     = (h16*)(ws + off); off += (size_t)Bn * Ln * CK * 2;  // 2 MB
  h16* gT      = (h16*)(ws + off); off += (size_t)Bn * CG * Ln * 2;  // 8 MB
  h16* attnpre = (h16*)(ws + off); off += (size_t)Bn * Ln * CG * 2;  // 8 MB
  h16* WoH     = (h16*)(ws + off); off += (size_t)Cn * CG * 2;       // 256 KB
  float* colmax = (float*)(ws + off); off += (size_t)Bn * Ln * 4;    // 64 KB
  float* colsum = (float*)(ws + off); off += (size_t)Bn * Ln * 4;    // 64 KB

  k_cvt_wo<<<(Cn * CG + 255) / 256, 256, 0, stream>>>(Wo, WoH);
  k_proj<<<dim3(Bn * (Ln / 64), 6), 256, 0, stream>>>(
      x, Wt, bt, Wp, bp, Wg, bg, theta, phi, gT);
  k_colstats<<<Bn * (Ln / 16), 128, 0, stream>>>(theta, phi, colmax, colsum);
  k_norm_g<<<(Bn * CG * Ln) / 256, 256, 0, stream>>>(gT, colsum);
  k_pv<<<dim3(Bn * (Ln / 64), 2), 128, 0, stream>>>(theta, phi, gT, colmax, attnpre);
  k_oproj<<<dim3(Bn * (Ln / 16), Cn / 64), 128, 0, stream>>>(
      WoH, attnpre, bo, gamma, x, out1, out2);
}
